// MACEModel_240518168928
// MI455X (gfx1250) — compile-verified, matrix operand order checked by reading, not run on hardware
//
#include <hip/hip_runtime.h>
#include <cmath>
#include <complex>

// ---------------- model constants ----------------
#define NPP     11
#define C_EMB   32
#define TILE_E  16
#define NBESS   8
#define RMAXF   10.0f
#define AVG_NEIGH 38.0f
#define CGTOT   363

typedef __attribute__((ext_vector_type(16))) _Float16 v16h;
typedef __attribute__((ext_vector_type(8)))  float    v8f;

// Path list (l1,l2,l3) in reference generator order; conv layer0 uses the first 3 (l1==0).
static const int hPP[NPP][3] = {
  {0,0,0},{0,1,1},{0,2,2},{1,0,1},{1,1,0},{1,1,2},{1,2,1},{2,0,2},{2,1,1},{2,2,0},{2,2,2}};
static const int hCGOFF[NPP] = {0,1,10,35,44,53,98,143,168,213,238};

__constant__ int dPP[NPP][3] = {
  {0,0,0},{0,1,1},{0,2,2},{1,0,1},{1,1,0},{1,1,2},{1,2,1},{2,0,2},{2,1,1},{2,2,0},{2,2,2}};
__constant__ int dCGOFF[NPP] = {0,1,10,35,44,53,98,143,168,213,238};

struct CGPack  { float v[CGTOT]; };
struct ConvArgs{ int npaths; float pscale[NPP]; };
struct ProdArgs{ float ppscale[NPP]; float invsqrtC; };

__device__ __forceinline__ int loff(int l){ return (l==0)?0:((l==1)?1:4); }

// ---------------- host: real Clebsch-Gordan (e3nn convention) ----------------
using cplx = std::complex<double>;
static double factd(int n){ double r=1.0; for(int i=2;i<=n;++i) r*=i; return r; }

static double su2_cg(int j1,int m1,int j2,int m2,int j3,int m3){
  if (m3 != m1+m2) return 0.0;
  int vmin = -j1+j2+m3; if (-j1+m1 > vmin) vmin = -j1+m1; if (0 > vmin) vmin = 0;
  int vmax = j2+j3+m1;  if (j3-j1+j2 < vmax) vmax = j3-j1+j2; if (j3+m3 < vmax) vmax = j3+m3;
  double pref = std::sqrt((2.0*j3+1.0) *
      (factd(j3+j1-j2)*factd(j3-j1+j2)*factd(j1+j2-j3)*factd(j3+m3)*factd(j3-m3)) /
      (factd(j1+j2+j3+1)*factd(j1-m1)*factd(j1+m1)*factd(j2-m2)*factd(j2+m2)));
  double s = 0.0;
  for (int v=vmin; v<=vmax; ++v){
    double sgn = ((v+j2+m2) & 1) ? -1.0 : 1.0;
    s += sgn * (factd(j2+j3+m1-v)*factd(j1-m1+v)) /
         (factd(v)*factd(j3-j1+j2-v)*factd(j3+m3-v)*factd(v+j1-j2-m3));
  }
  return pref*s;
}

static void q_r2c(int l, cplx q[5][5]){
  for (int a=0;a<5;++a) for (int b=0;b<5;++b) q[a][b] = cplx(0,0);
  const double s2 = 1.0/std::sqrt(2.0);
  for (int m=-l; m<0; ++m){
    q[l+m][l-m] = cplx(s2,0.0);   // column l+|m|
    q[l+m][l+m] = cplx(0.0,-s2);  // column l-|m|
  }
  q[l][l] = cplx(1.0,0.0);
  for (int m=1; m<=l; ++m){
    double sgn = (m&1)? -1.0 : 1.0;
    q[l+m][l+m] = cplx(sgn*s2,0.0);
    q[l+m][l-m] = cplx(0.0,sgn*s2);
  }
  cplx f = (l==0)?cplx(1,0):((l==1)?cplx(0,-1):cplx(-1,0)); // (-i)^l
  for (int a=0;a<2*l+1;++a) for (int b=0;b<2*l+1;++b) q[a][b] *= f;
}

static void real_cg(int l1,int l2,int l3, float* out){
  int d1=2*l1+1, d2=2*l2+1, d3=2*l3+1;
  double c[5][5][5] = {};
  for (int m1=-l1;m1<=l1;++m1) for (int m2=-l2;m2<=l2;++m2){
    int m3 = m1+m2;
    if (m3 >= -l3 && m3 <= l3) c[l1+m1][l2+m2][l3+m3] = su2_cg(l1,m1,l2,m2,l3,m3);
  }
  cplx q1[5][5], q2[5][5], q3[5][5];
  q_r2c(l1,q1); q_r2c(l2,q2); q_r2c(l3,q3);
  for (int j=0;j<d1;++j) for (int l=0;l<d2;++l) for (int m=0;m<d3;++m){
    cplx s(0,0);
    for (int i=0;i<d1;++i) for (int k=0;k<d2;++k) for (int n=0;n<d3;++n)
      s += q1[i][j]*q2[k][l]*std::conj(q3[n][m])*c[i][k][n];
    out[(j*d2+l)*d3+m] = (float)s.real();
  }
}

// ---------------- kernels ----------------

__global__ void zero_kernel(float* __restrict__ p, int n){
  int i = blockIdx.x*blockDim.x + threadIdx.x;
  if (i < n) p[i] = 0.f;
}

// h0 embedding: h[n][c][0] = z*W0 + t*W1 + b ; l>0 comps = 0
__global__ void embed_kernel(const float* __restrict__ z, const float* __restrict__ tf,
                             const float* __restrict__ embW, const float* __restrict__ embB,
                             float* __restrict__ h, int N){
  int i = blockIdx.x*blockDim.x + threadIdx.x;
  if (i >= N*C_EMB) return;
  int n = i / C_EMB, c = i % C_EMB;
  float v = z[n]*embW[c] + tf[n]*embW[C_EMB+c] + embB[c];
  float* hp = h + (size_t)i*9;
  hp[0] = v;
  #pragma unroll
  for (int k=1;k<9;++k) hp[k] = 0.f;
}

// per-edge: spherical harmonics (via CG, matching reference), bessel radial, fc1+silu (f16)
__global__ void edgeprep_kernel(const float* __restrict__ evec,
    const float* __restrict__ fc1W0, const float* __restrict__ fc1b0,
    const float* __restrict__ fc1W1, const float* __restrict__ fc1b1,
    float* __restrict__ sh, _Float16* __restrict__ hid, CGPack cg, int E){
  int e = blockIdx.x*blockDim.x + threadIdx.x;
  if (e >= E) return;
  float x = evec[3*e], y = evec[3*e+1], z = evec[3*e+2];
  float r = sqrtf(x*x + y*y + z*z);
  float inv = 1.f / r;
  float nx = x*inv, ny = y*inv, nz = z*inv;
  const float S3 = 1.7320508075688772f;
  float y1v[3] = { S3*ny, S3*nz, S3*nx };
  float y2v[5];
  const float* cg112 = cg.v + 53;  // CG(1,1,2)
  float fac = sqrtf(5.f/6.f);
  #pragma unroll
  for (int k=0;k<5;++k){
    float s = 0.f;
    for (int i=0;i<3;++i) for (int j=0;j<3;++j)
      s += cg112[(i*3+j)*5 + k]*y1v[i]*y1v[j];
    y2v[k] = fac*s;
  }
  float* shp = sh + (size_t)e*9;
  shp[0] = 1.f;
  #pragma unroll
  for (int i=0;i<3;++i) shp[1+i] = y1v[i];
  #pragma unroll
  for (int i=0;i<5;++i) shp[4+i] = y2v[i];
  // Bessel * polynomial cutoff (p=5)
  float u = r / RMAXF;
  float env = 0.f;
  if (u < 1.f){
    float u2=u*u, u4=u2*u2, u5=u4*u, u6=u5*u, u7=u6*u;
    env = 1.f - 21.f*u5 + 35.f*u6 - 15.f*u7;
  }
  float ef[NBESS];
  const float PIF = 3.14159265358979f;
  float pref = sqrtf(2.f/RMAXF);
  #pragma unroll
  for (int nb=0;nb<NBESS;++nb)
    ef[nb] = pref*sinf((float)(nb+1)*PIF*r/RMAXF)*inv*env;
  const float* W[2] = { fc1W0, fc1W1 };
  const float* B[2] = { fc1b0, fc1b1 };
  for (int L=0;L<2;++L){
    _Float16* hp = hid + ((size_t)L*E + e)*C_EMB;
    for (int c=0;c<C_EMB;++c){
      float a = B[L][c];
      #pragma unroll
      for (int nb=0;nb<NBESS;++nb) a += ef[nb]*W[L][nb*C_EMB+c];
      a = a / (1.f + expf(-a));       // silu
      hp[c] = (_Float16)a;
    }
  }
}

// Pack fc2_W (both layers) into WMMA B-fragment layout (f16):
// per 16-col tile T: frag[(T*32+lane)*16+kk]; lanes 0-15 hold K{0..7,16..23}, lanes 16-31 K{8..15,24..31}.
__global__ void fragprep_kernel(const float* __restrict__ fc2W0,
                                const float* __restrict__ fc2W1,
                                _Float16* __restrict__ frag){
  const int H0 = 3072*C_EMB;            // layer0 halves
  const int HT = H0 + 11264*C_EMB;      // total halves
  int idx = blockIdx.x*blockDim.x + threadIdx.x;
  if (idx >= HT) return;
  const float* W; int numel; int local;
  if (idx < H0){ W = fc2W0; numel = 3072;  local = idx; }
  else         { W = fc2W1; numel = 11264; local = idx - H0; }
  int tile = local >> 9;        // /512 halves per tile
  int rem  = local & 511;
  int lane = rem >> 4;
  int kk   = rem & 15;
  int n    = lane & 15;
  int k    = (lane < 16) ? ((kk < 8) ? kk : kk + 8)
                         : ((kk < 8) ? kk + 8 : kk + 16);
  int col  = tile*16 + n;
  frag[idx] = (_Float16)W[(size_t)k*numel + col];
}

// Fused per-edge weight-generation (WMMA) + tensor product + scatter.
__global__ __launch_bounds__(256) void conv_kernel(
    const float*    __restrict__ hprev,   // [N][32][9]
    const _Float16* __restrict__ hid16,   // [E][32] (layer slice)
    const _Float16* __restrict__ fc2frag, // B-fragment layout (layer slice)
    const float*    __restrict__ fc2b,    // [numel]
    const float*    __restrict__ sh,      // [E][9]
    const int*      __restrict__ eidx,    // [2][E]
    float*          __restrict__ msgs,    // [N][32][9]
    CGPack cg, ConvArgs args, int E){
  __shared__ float Wlds[TILE_E][520];         // half of the 1024 weight cols (+pad)
  __shared__ float hsrc[TILE_E][C_EMB][9];
  __shared__ float tl[TILE_E][C_EMB][5];
  __shared__ float shl[TILE_E][9];
  __shared__ int   dstl[TILE_E];

  int tid = threadIdx.x;
  int e0  = blockIdx.x * TILE_E;

  for (int i = tid; i < TILE_E*9; i += 256){
    int eg = e0 + i/9; if (eg >= E) eg = E-1;
    shl[i/9][i%9] = sh[(size_t)eg*9 + i%9];
  }
  if (tid < TILE_E){
    int eg = e0 + tid; if (eg >= E) eg = E-1;
    dstl[tid] = eidx[E + eg];
  }
  for (int i = tid; i < TILE_E*C_EMB*9; i += 256){
    int e = i / (C_EMB*9); int rem = i % (C_EMB*9);
    int eg = e0 + e; if (eg >= E) eg = E-1;
    int s = eidx[eg];
    hsrc[e][rem/9][rem%9] = hprev[(size_t)s*(C_EMB*9) + rem];
  }
  __syncthreads();

  int lane = tid & 31;
  int wave = tid >> 5;
  // A-fragment: M = edge-in-tile, K = hidden channel (f16, ISA 16-bit A layout)
  v16h afrag;
  {
    int m  = lane & 15;
    int k0 = (lane < 16) ? 0 : 8;
    int eg = e0 + m; if (eg >= E) eg = E-1;
    const _Float16* hp = hid16 + (size_t)eg * C_EMB;
    #pragma unroll
    for (int kk=0;kk<8;++kk){ afrag[kk] = hp[k0+kk]; afrag[kk+8] = hp[k0+16+kk]; }
  }

  float acc[2][9];
  #pragma unroll
  for (int p=0;p<2;++p)
    #pragma unroll
    for (int k=0;k<9;++k) acc[p][k] = 0.f;

  for (int pidx = 0; pidx < args.npaths; ++pidx){
    int l1 = dPP[pidx][0], l2 = dPP[pidx][1], l3 = dPP[pidx][2];
    int d1 = 2*l1+1, d2 = 2*l2+1, d3 = 2*l3+1;
    int o1 = loff(l1), o2 = loff(l2), o3 = loff(l3);
    const float* cgp = cg.v + dCGOFF[pidx];

    __syncthreads();  // previous apply done (tl / Wlds reusable)
    // t[e,u,k] = sum_ij h_src[e,u,i] * sh[e,j] * CG[i,j,k]
    for (int pair = tid; pair < TILE_E*C_EMB; pair += 256){
      int e = pair >> 5, u = pair & 31;
      float tv[5];
      for (int k=0;k<d3;++k) tv[k] = 0.f;
      for (int i=0;i<d1;++i){
        float hv = hsrc[e][u][o1+i];
        for (int j=0;j<d2;++j){
          float hs = hv * shl[e][o2+j];
          for (int k=0;k<d3;++k) tv[k] += hs * cgp[(i*d2+j)*d3 + k];
        }
      }
      for (int k=0;k<d3;++k) tl[e][u][k] = tv[k];
    }

    size_t ewoff = (size_t)pidx * (C_EMB*C_EMB);
    float scale  = args.pscale[pidx];

    for (int half = 0; half < 2; ++half){
      __syncthreads();  // t visible / previous half's apply done
      // weight block: 16 edges x 512 cols via 4 WMMA tiles per wave
      #pragma unroll
      for (int i = 0; i < 4; ++i){
        int nt = half*32 + wave*4 + i;              // 0..63 column tile within path
        size_t Tg = (ewoff >> 4) + (size_t)nt;      // global column tile in fc2frag
        v16h bfrag = *(const v16h*)(fc2frag + (Tg*32 + (size_t)lane)*16);
        v8f cacc;
        float bias = fc2b[ewoff + (size_t)nt*16 + (lane & 15)];
        #pragma unroll
        for (int r=0;r<8;++r) cacc[r] = bias;
        cacc = __builtin_amdgcn_wmma_f32_16x16x32_f16(
                   false, afrag, false, bfrag, (short)0, cacc, false, false);
        int mbase = (lane < 16) ? 0 : 8;
        int col   = (nt - half*32)*16 + (lane & 15);
        #pragma unroll
        for (int r=0;r<8;++r) Wlds[mbase + r][col] = cacc[r];
      }
      __syncthreads();  // W visible
      // apply: m[e,w,k] += scale * sum_u W[e,u,w]*t[e,u,k]  (u in this half)
      int u0 = half*16;
      #pragma unroll
      for (int pp = 0; pp < 2; ++pp){
        int pair = tid + pp*256;
        int e = pair >> 5, w = pair & 31;
        for (int k=0;k<d3;++k){
          float s = 0.f;
          #pragma unroll
          for (int uu=0; uu<16; ++uu)
            s += Wlds[e][uu*C_EMB + w] * tl[e][u0+uu][k];
          acc[pp][o3+k] += scale * s;
        }
      }
    }
  }
  __syncthreads();
  const float inv_avg = 1.0f / AVG_NEIGH;
  #pragma unroll
  for (int pp=0;pp<2;++pp){
    int pair = tid + pp*256;
    int e = pair >> 5, w = pair & 31;
    if (e0 + e < E){
      int d = dstl[e];
      #pragma unroll
      for (int k=0;k<9;++k)
        atomicAdd(&msgs[((size_t)d*C_EMB + w)*9 + k], acc[pp][k]*inv_avg);
    }
  }
}

// symmetric-contraction helper: out[l3] += w[p]*ppscale[p] * sum_ij B[l1,i]*A[l2,j]*CG
__device__ __forceinline__ void contract_dev(const float* Bv, const float* Av,
    const float* __restrict__ wmat, int c, const float* cgv,
    const float* ppscale, float* outv){
  for (int p=0;p<NPP;++p){
    int l1 = dPP[p][0], l2 = dPP[p][1], l3 = dPP[p][2];
    int d1 = 2*l1+1, d2 = 2*l2+1, d3 = 2*l3+1;
    int o1 = loff(l1), o2 = loff(l2), o3 = loff(l3);
    const float* cgp = cgv + dCGOFF[p];
    float wv = wmat[p*C_EMB + c] * ppscale[p];
    for (int k=0;k<d3;++k){
      float t = 0.f;
      for (int i=0;i<d1;++i) for (int j=0;j<d2;++j)
        t += Bv[o1+i]*Av[o2+j]*cgp[(i*d2+j)*d3 + k];
      outv[o3+k] += wv*t;
    }
  }
}

// EquivariantProductBasisBlock: B2/B3 correlations + per-l channel-mixing linear + skip
__global__ __launch_bounds__(256) void prod_kernel(
    const float* __restrict__ msgs, const float* __restrict__ hprev,
    const float* __restrict__ w1,  const float* __restrict__ w2,
    const float* __restrict__ w3,  const float* __restrict__ lin,
    float* __restrict__ hout, CGPack cg, ProdArgs pa, int N){
  __shared__ float ml[8][C_EMB][9];
  int tid  = threadIdx.x;
  int nloc = tid >> 5, c = tid & 31;
  int n  = blockIdx.x*8 + nloc;
  int nc = (n < N) ? n : (N-1);
  float A9[9], B2v[9], B3v[9];
  const float* ap = msgs + ((size_t)nc*C_EMB + c)*9;
  #pragma unroll
  for (int k=0;k<9;++k){ A9[k] = ap[k]; B2v[k] = 0.f; B3v[k] = 0.f; }
  contract_dev(A9,  A9, w2, c, cg.v, pa.ppscale, B2v);
  contract_dev(B2v, A9, w3, c, cg.v, pa.ppscale, B3v);
  #pragma unroll
  for (int l=0;l<3;++l){
    int o = loff(l), d = 2*l+1;
    float wv = w1[l*C_EMB + c];
    for (int k=0;k<d;++k) ml[nloc][c][o+k] = wv*A9[o+k] + B2v[o+k] + B3v[o+k];
  }
  __syncthreads();
  if (n < N){
    for (int k=0;k<9;++k){
      int l = (k==0)?0:((k<4)?1:2);
      float s = 0.f;
      for (int cc=0; cc<C_EMB; ++cc)
        s += ml[nloc][cc][k] * lin[((size_t)l*C_EMB + cc)*C_EMB + c];
      hout[((size_t)n*C_EMB + c)*9 + k] =
          s*pa.invsqrtC + hprev[((size_t)n*C_EMB + c)*9 + k];
    }
  }
}

// readout: softplus MLP on l=0 channels
__global__ void readout_kernel(const float* __restrict__ h,
    const float* __restrict__ W1, const float* __restrict__ b1,
    const float* __restrict__ W2, const float* __restrict__ b2,
    float* __restrict__ out, int N){
  int n = blockIdx.x*blockDim.x + threadIdx.x;
  if (n >= N) return;
  float s[C_EMB];
  #pragma unroll
  for (int c=0;c<C_EMB;++c) s[c] = h[((size_t)n*C_EMB + c)*9];
  float o = b2[0];
  for (int d=0; d<C_EMB; ++d){
    float a = b1[d];
    #pragma unroll
    for (int c=0;c<C_EMB;++c) a += s[c]*W1[c*C_EMB + d];
    a = (a > 20.f) ? a : log1pf(expf(a));   // softplus
    o += a*W2[d];
  }
  out[n] = o;
}

// ---------------- launch ----------------
extern "C" void kernel_launch(void* const* d_in, const int* in_sizes, int n_in,
                              void* d_out, int out_size, void* d_ws, size_t ws_size,
                              hipStream_t stream){
  (void)n_in; (void)out_size; (void)ws_size;
  const float* z    = (const float*)d_in[0];
  const float* tf   = (const float*)d_in[1];
  const int*   eidx = (const int*)  d_in[2];
  const float* evec = (const float*)d_in[3];
  const float* embW = (const float*)d_in[4];
  const float* embB = (const float*)d_in[5];
  const float* fc1W[2] = {(const float*)d_in[6],  (const float*)d_in[14]};
  const float* fc1b[2] = {(const float*)d_in[7],  (const float*)d_in[15]};
  const float* fc2W[2] = {(const float*)d_in[8],  (const float*)d_in[16]};
  const float* fc2b[2] = {(const float*)d_in[9],  (const float*)d_in[17]};
  const float* w1p[2]  = {(const float*)d_in[10], (const float*)d_in[18]};
  const float* w2p[2]  = {(const float*)d_in[11], (const float*)d_in[19]};
  const float* w3p[2]  = {(const float*)d_in[12], (const float*)d_in[20]};
  const float* linp[2] = {(const float*)d_in[13], (const float*)d_in[21]};
  const float* pW1 = (const float*)d_in[22];
  const float* pb1 = (const float*)d_in[23];
  const float* pW2 = (const float*)d_in[24];
  const float* pb2 = (const float*)d_in[25];
  float* out = (float*)d_out;

  const int N = in_sizes[0];
  const int E = in_sizes[2] / 2;

  // host-computed real CG tables (deterministic)
  CGPack cg;
  for (int p=0;p<NPP;++p)
    real_cg(hPP[p][0], hPP[p][1], hPP[p][2], cg.v + hCGOFF[p]);

  // per-layer conv args: npaths + 1/sqrt(C*fan[l3])
  ConvArgs ca[2];
  int npl[2] = {3, NPP};
  for (int L=0;L<2;++L){
    ca[L].npaths = npl[L];
    int fan[3] = {0,0,0};
    for (int p=0;p<npl[L];++p) fan[hPP[p][2]]++;
    for (int p=0;p<npl[L];++p)
      ca[L].pscale[p] = 1.0f / std::sqrt((float)C_EMB * (float)fan[hPP[p][2]]);
    for (int p=npl[L];p<NPP;++p) ca[L].pscale[p] = 0.f;
  }
  ProdArgs pa;
  {
    int pfan[3] = {0,0,0};
    for (int p=0;p<NPP;++p) pfan[hPP[p][2]]++;
    for (int p=0;p<NPP;++p) pa.ppscale[p] = 1.0f / std::sqrt((float)pfan[hPP[p][2]]);
    pa.invsqrtC = 1.0f / std::sqrt((float)C_EMB);
  }

  // workspace carve (256B aligned regions)
  char* base = (char*)d_ws;
  size_t off = 0;
  auto carve = [&](size_t bytes)->char*{
    char* p = base + off;
    off = (off + bytes + 255) & ~(size_t)255;
    return p;
  };
  float* hA   = (float*)carve((size_t)N*C_EMB*9*sizeof(float));
  float* hB   = (float*)carve((size_t)N*C_EMB*9*sizeof(float));
  float* msgs = (float*)carve((size_t)N*C_EMB*9*sizeof(float));
  float* shw  = (float*)carve((size_t)E*9*sizeof(float));
  _Float16* hid  = (_Float16*)carve((size_t)2*E*C_EMB*sizeof(_Float16));
  _Float16* frag = (_Float16*)carve((size_t)(3072+11264)*C_EMB*sizeof(_Float16));

  const int msz = N*C_EMB*9;
  // prep
  embed_kernel<<<(N*C_EMB + 255)/256, 256, 0, stream>>>(z, tf, embW, embB, hA, N);
  edgeprep_kernel<<<(E + 255)/256, 256, 0, stream>>>(
      evec, fc1W[0], fc1b[0], fc1W[1], fc1b[1], shw, hid, cg, E);
  {
    int tot = (3072 + 11264)*C_EMB;
    fragprep_kernel<<<(tot + 255)/256, 256, 0, stream>>>(fc2W[0], fc2W[1], frag);
  }

  float* hin  = hA;
  float* hnew = hB;
  for (int L=0; L<2; ++L){
    zero_kernel<<<(msz + 255)/256, 256, 0, stream>>>(msgs, msz);
    const _Float16* fragL = frag + (L==0 ? 0 : (size_t)3072*C_EMB);
    conv_kernel<<<(E + TILE_E - 1)/TILE_E, 256, 0, stream>>>(
        hin, hid + (size_t)L*E*C_EMB, fragL, fc2b[L], shw, eidx, msgs, cg, ca[L], E);
    prod_kernel<<<(N + 7)/8, 256, 0, stream>>>(
        msgs, hin, w1p[L], w2p[L], w3p[L], linp[L], hnew, cg, pa, N);
    float* tmp = hin; hin = hnew; hnew = tmp;
  }
  readout_kernel<<<(N + 255)/256, 256, 0, stream>>>(hin, pW1, pb1, pW2, pb2, out, N);
}
